// MultiLatentAttention_19421842113028
// MI455X (gfx1250) — matrix operand order repaired and checked by reading, not compiled, not run on hardware
//
#include <hip/hip_runtime.h>
#include <hip/hip_bf16.h>

// ---------------------------------------------------------------------------
// MLA forward for MI455X (gfx1250): all matmuls via v_wmma_f32_16x16x32_f16.
// GEMM LDS staging via GLOBAL_LOAD_ASYNC_TO_LDS_B128 (ASYNCcnt-tracked).
// ---------------------------------------------------------------------------

typedef __attribute__((ext_vector_type(16))) _Float16 v16h;
typedef __attribute__((ext_vector_type(8)))  _Float16 v8h;
typedef __attribute__((ext_vector_type(8)))  float    v8f;
typedef __attribute__((ext_vector_type(4)))  int      v4i;

#define EMBED_DIM 2048
#define NUM_HEADS 16
#define Q_LORA    512
#define KV_LORA   512
#define ROPE_DIM  64
#define NOPE_DIM  128
#define V_DIM     128
#define QH_DIM    192   // NOPE + ROPE

#if __has_builtin(__builtin_amdgcn_global_load_async_to_lds_b128) && \
    __has_builtin(__builtin_amdgcn_s_wait_asynccnt)
#define USE_ASYNC_LDS 1
typedef __attribute__((address_space(1))) v4i* gv4i_p;
typedef __attribute__((address_space(3))) v4i* lv4i_p;
#else
#define USE_ASYNC_LDS 0
#endif

__device__ __forceinline__ v8f wmma_f16(v16h a, v16h b, v8f c) {
  return __builtin_amdgcn_wmma_f32_16x16x32_f16(
      /*neg_a=*/false, a, /*neg_b=*/false, b,
      /*c_mod=*/(short)0, c, /*reuse_a=*/false, /*reuse_b=*/false);
}

// A-matrix 16x32 f16 fragment (ISA §7.12.2, 16-bit A):
// lanes 0-15 -> M=lane, K {0..7, 16..23}; lanes 16-31 -> M=lane-16, K {8..15, 24..31}.
__device__ __forceinline__ v16h load_frag_a16x32(const _Float16* base, int ld) {
  const int lane = threadIdx.x & 31;
  const int r    = lane & 15;
  const int koff = (lane & 16) ? 8 : 0;
  const _Float16* p = base + (size_t)r * ld + koff;
  v8h lo = *(const v8h*)(p);        // K = koff .. koff+7
  v8h hi = *(const v8h*)(p + 16);   // K = koff+16 .. koff+23
  v16h f;
#pragma unroll
  for (int i = 0; i < 8; ++i) { f[i] = lo[i]; f[i + 8] = hi[i]; }
  return f;
}

// B-matrix 32x16 f16 fragment, source stored transposed (Bt[N][K], K contiguous).
// K order matches the A fragment:
// lanes 0-15 -> N=lane, K {0..7, 16..23}; lanes 16-31 -> N=lane-16, K {8..15, 24..31}.
__device__ __forceinline__ v16h load_frag_b16x32(const _Float16* base, int ld) {
  const int lane = threadIdx.x & 31;
  const int r    = lane & 15;
  const int koff = (lane & 16) ? 8 : 0;
  const _Float16* p = base + (size_t)r * ld + koff;
  v8h lo = *(const v8h*)(p);        // K = koff .. koff+7
  v8h hi = *(const v8h*)(p + 16);   // K = koff+16 .. koff+23
  v16h f;
#pragma unroll
  for (int i = 0; i < 8; ++i) { f[i] = lo[i]; f[i + 8] = hi[i]; }
  return f;
}

// ---------------------------------------------------------------------------
// Generic f16 WMMA GEMM: C[M,N] = A[M,K] * B[K,N], with B supplied TRANSPOSED
// (Bt[N,K]) so fragments load K-contiguous. Block tile 128x64, 8 waves (4x2),
// wave tile 32x32 (2x2 WMMA tiles), K step 32. Requires M%128==N%64==K%32==0.
// ---------------------------------------------------------------------------
template <bool OUT_HALF>
__global__ __launch_bounds__(256)
void gemm_wmma(const _Float16* __restrict__ A, int lda,
               const _Float16* __restrict__ Bt, int ldb,
               void* __restrict__ Cout, int ldc, int K) {
  __shared__ _Float16 As[128][40];   // pad row stride to 80B (16B aligned)
  __shared__ _Float16 Bs[64][40];

  const int m0 = blockIdx.x * 128;
  const int n0 = blockIdx.y * 64;
  const int t  = threadIdx.x;
  const int w  = t >> 5;
  const int wm = w >> 1;                 // 0..3
  const int wn = w & 1;                  // 0..1
  const int arow = t >> 1, aseg = (t & 1) * 16;  // 2 thr/row, 16 halves each
  const int brow = t >> 2, bseg = (t & 3) * 8;   // 4 thr/row, 8 halves each

  v8f acc[2][2] = {};

  for (int k0 = 0; k0 < K; k0 += 32) {
    const _Float16* ga = A  + (size_t)(m0 + arow) * lda + k0 + aseg;
    const _Float16* gb = Bt + (size_t)(n0 + brow) * ldb + k0 + bseg;
#if USE_ASYNC_LDS
    __builtin_amdgcn_global_load_async_to_lds_b128(
        (gv4i_p)(ga), (lv4i_p)(&As[arow][aseg]), 0, 0);
    __builtin_amdgcn_global_load_async_to_lds_b128(
        (gv4i_p)(ga + 8), (lv4i_p)(&As[arow][aseg + 8]), 0, 0);
    __builtin_amdgcn_global_load_async_to_lds_b128(
        (gv4i_p)(gb), (lv4i_p)(&Bs[brow][bseg]), 0, 0);
    __builtin_amdgcn_s_wait_asynccnt(0);
#else
    *(v8h*)&As[arow][aseg]     = *(const v8h*)(ga);
    *(v8h*)&As[arow][aseg + 8] = *(const v8h*)(ga + 8);
    *(v8h*)&Bs[brow][bseg]     = *(const v8h*)(gb);
#endif
    // hint next K tile into cache (global_prefetch_b8)
    __builtin_prefetch(ga + 32, 0, 3);
    __builtin_prefetch(gb + 32, 0, 3);
    __syncthreads();

    v16h af0 = load_frag_a16x32(&As[wm * 32][0],      40);
    v16h af1 = load_frag_a16x32(&As[wm * 32 + 16][0], 40);
    v16h bf0 = load_frag_b16x32(&Bs[wn * 32][0],      40);
    v16h bf1 = load_frag_b16x32(&Bs[wn * 32 + 16][0], 40);
    acc[0][0] = wmma_f16(af0, bf0, acc[0][0]);
    acc[0][1] = wmma_f16(af0, bf1, acc[0][1]);
    acc[1][0] = wmma_f16(af1, bf0, acc[1][0]);
    acc[1][1] = wmma_f16(af1, bf1, acc[1][1]);
    __syncthreads();
  }

  // C/D layout: lanes 0-15 -> N=lane, rows M=vgpr; lanes 16-31 -> rows M=8+vgpr.
  const int lane = t & 31;
  const int cr = (lane & 16) ? 8 : 0;
  const int cc = lane & 15;
#pragma unroll
  for (int i = 0; i < 2; ++i)
#pragma unroll
    for (int j = 0; j < 2; ++j)
#pragma unroll
      for (int v = 0; v < 8; ++v) {
        const size_t m = (size_t)m0 + wm * 32 + i * 16 + cr + v;
        const size_t n = (size_t)n0 + wn * 32 + j * 16 + cc;
        if (OUT_HALF) ((_Float16*)Cout)[m * ldc + n] = (_Float16)acc[i][j][v];
        else          ((float*)Cout)[m * ldc + n]    = acc[i][j][v];
      }
}

// ---------------------------------------------------------------------------
// Conversions
// ---------------------------------------------------------------------------
__global__ void cvt_f32_f16(const float* __restrict__ in,
                            _Float16* __restrict__ out, int n) {
  int i = blockIdx.x * 256 + threadIdx.x;
  if (i < n) out[i] = (_Float16)in[i];
}

// W[K][N] fp32 -> Wt[N][K] f16
__global__ void cvt_transpose(const float* __restrict__ W,
                              _Float16* __restrict__ Wt, int K, int N) {
  int i = blockIdx.x * 256 + threadIdx.x;
  if (i < K * N) {
    int k = i / N, n = i - k * N;
    Wt[(size_t)n * K + k] = (_Float16)W[i];
  }
}

// ---------------------------------------------------------------------------
// RoPE (interleaved pairs), in place on f16 buffers; trig in fp32.
// ---------------------------------------------------------------------------
__global__ void rope_q(_Float16* __restrict__ q, int total, int nseq) {
  int i = blockIdx.x * 256 + threadIdx.x;
  if (i >= total) return;                  // total = B*N*H*32
  int pair = i & 31;
  int h    = (i >> 5) & 15;
  int bn   = i >> 9;
  int pos  = bn % nseq;
  float inv = __expf(-((float)(2 * pair) / (float)ROPE_DIM) * __logf(10000.0f));
  float s, c;
  __sincosf((float)pos * inv, &s, &c);
  _Float16* p = q + ((size_t)bn * NUM_HEADS + h) * QH_DIM + NOPE_DIM + pair * 2;
  float x1 = (float)p[0], x2 = (float)p[1];
  p[0] = (_Float16)(x1 * c - x2 * s);
  p[1] = (_Float16)(x1 * s + x2 * c);
}

__global__ void rope_k(_Float16* __restrict__ ckv, int total, int nseq) {
  int i = blockIdx.x * 256 + threadIdx.x;
  if (i >= total) return;                  // total = B*N*32
  int pair = i & 31;
  int bn   = i >> 5;
  int pos  = bn % nseq;
  float inv = __expf(-((float)(2 * pair) / (float)ROPE_DIM) * __logf(10000.0f));
  float s, c;
  __sincosf((float)pos * inv, &s, &c);
  _Float16* p = ckv + (size_t)bn * (KV_LORA + ROPE_DIM) + KV_LORA + pair * 2;
  float x1 = (float)p[0], x2 = (float)p[1];
  p[0] = (_Float16)(x1 * c - x2 * s);
  p[1] = (_Float16)(x1 * s + x2 * c);
}

// ---------------------------------------------------------------------------
// Pack per-head K (kf[b,h,n,192] = k_nope | roped k_pe) and V transposed
// (vt[b,h,d,n]) so attention B fragments are K-contiguous row loads.
// grid = (nseq, H, B), block = 192
// ---------------------------------------------------------------------------
__global__ void pack_kv(const _Float16* __restrict__ kv,
                        const _Float16* __restrict__ ckv,
                        _Float16* __restrict__ kf,
                        _Float16* __restrict__ vt, int nseq) {
  const int n = blockIdx.x, h = blockIdx.y, b = blockIdx.z, d = threadIdx.x;
  const size_t bn = (size_t)b * nseq + n;
  const size_t bh = (size_t)b * NUM_HEADS + h;
  if (d < NOPE_DIM) {
    kf[(bh * nseq + n) * QH_DIM + d] = kv[bn * 4096 + h * 256 + d];
    vt[(bh * 128 + d) * nseq + n]    = kv[bn * 4096 + h * 256 + 128 + d];
  } else {
    kf[(bh * nseq + n) * QH_DIM + d] =
        ckv[bn * (KV_LORA + ROPE_DIM) + KV_LORA + (d - NOPE_DIM)];
  }
}

// ---------------------------------------------------------------------------
// Causal flash attention. One wave owns 16 query rows and the full key loop.
// Per 32-key tile: 12 WMMAs (S = Q*K^T, d=192) + 8 WMMAs (O += P*V^T).
// Online softmax on the C layout: row j lives in VGPR j across a 16-lane half.
// grid = (nseq/128, H, B), block = 256 (8 waves).
// ---------------------------------------------------------------------------
__global__ __launch_bounds__(256)
void mla_attn(const _Float16* __restrict__ q,    // [b,n,h,192]
              const _Float16* __restrict__ kf,   // [b,h,n,192]
              const _Float16* __restrict__ vt,   // [b,h,128,n]
              _Float16* __restrict__ ao,         // [b,n,h*128]
              int nseq) {
  __shared__ _Float16 Ps[8][16][40];             // wave-private P tiles

  const int w = threadIdx.x >> 5, lane = threadIdx.x & 31;
  const int h = blockIdx.y, b = blockIdx.z;
  const int rbase = blockIdx.x * 128 + w * 16;
  const size_t bh = (size_t)b * NUM_HEADS + h;
  const int cr = (lane & 16) ? 8 : 0, cc = lane & 15;
  const float sc = rsqrtf((float)QH_DIM);

  // Q fragments: 16 rows x 192, row stride H*192 (layout [b,n,h,192])
  v16h qf[6];
  const _Float16* qp = q + (((size_t)b * nseq + rbase) * NUM_HEADS + h) * QH_DIM;
#pragma unroll
  for (int c = 0; c < 6; ++c)
    qf[c] = load_frag_a16x32(qp + c * 32, NUM_HEADS * QH_DIM);

  v8f o[8] = {};
  float m[8], l[8];
#pragma unroll
  for (int j = 0; j < 8; ++j) { m[j] = -1e30f; l[j] = 0.0f; }

  const int nkt = (rbase + 47) >> 5;             // causal key-tile count
  for (int kt = 0; kt < nkt; ++kt) {
    const int k0 = kt * 32;
    const _Float16* kp = kf + (bh * nseq + k0) * QH_DIM;
    v8f s0 = {}, s1 = {};
#pragma unroll
    for (int c = 0; c < 6; ++c) {
      v16h kb0 = load_frag_b16x32(kp + c * 32,                 QH_DIM);
      v16h kb1 = load_frag_b16x32(kp + 16 * QH_DIM + c * 32,   QH_DIM);
      s0 = wmma_f16(qf[c], kb0, s0);
      s1 = wmma_f16(qf[c], kb1, s1);
    }
#pragma unroll
    for (int j = 0; j < 8; ++j) {
      const int row = rbase + cr + j;
      float a0 = (k0 + cc      <= row) ? s0[j] * sc : -1e30f;
      float a1 = (k0 + 16 + cc <= row) ? s1[j] * sc : -1e30f;
      float t = fmaxf(a0, a1);                    // row max over 16 lanes
      t = fmaxf(t, __shfl_xor(t, 1));
      t = fmaxf(t, __shfl_xor(t, 2));
      t = fmaxf(t, __shfl_xor(t, 4));
      t = fmaxf(t, __shfl_xor(t, 8));
      const float mn    = fmaxf(m[j], t);
      const float scale = __expf(m[j] - mn);
      const float p0 = __expf(a0 - mn);
      const float p1 = __expf(a1 - mn);
      float rs = p0 + p1;
      rs += __shfl_xor(rs, 1);
      rs += __shfl_xor(rs, 2);
      rs += __shfl_xor(rs, 4);
      rs += __shfl_xor(rs, 8);
      l[j] = l[j] * scale + rs;
      m[j] = mn;
#pragma unroll
      for (int t2 = 0; t2 < 8; ++t2) o[t2][j] *= scale;
      Ps[w][cr + j][cc]      = (_Float16)p0;      // relayout C -> A via LDS
      Ps[w][cr + j][16 + cc] = (_Float16)p1;
    }
    v16h pf = load_frag_a16x32(&Ps[w][0][0], 40); // P: 16 rows x 32 keys
    const _Float16* vp = vt + (bh * 128) * nseq + k0;
#pragma unroll
    for (int t2 = 0; t2 < 8; ++t2) {
      v16h vb = load_frag_b16x32(vp + (size_t)(t2 * 16) * nseq, nseq);
      o[t2] = wmma_f16(pf, vb, o[t2]);
    }
  }

#pragma unroll
  for (int j = 0; j < 8; ++j) {
    const float inv = 1.0f / l[j];
    const int row = rbase + cr + j;
    _Float16* op = ao + ((size_t)b * nseq + row) * EMBED_DIM + h * V_DIM;
#pragma unroll
    for (int t2 = 0; t2 < 8; ++t2)
      op[t2 * 16 + cc] = (_Float16)(o[t2][j] * inv);
  }
}

// ---------------------------------------------------------------------------
// Host-side orchestration
// ---------------------------------------------------------------------------
extern "C" void kernel_launch(void* const* d_in, const int* in_sizes, int n_in,
                              void* d_out, int out_size, void* d_ws,
                              size_t ws_size, hipStream_t stream) {
  (void)in_sizes; (void)n_in; (void)out_size; (void)ws_size;
  const float* x    = (const float*)d_in[0];
  const float* Wqa  = (const float*)d_in[1];
  const float* Wqb  = (const float*)d_in[2];
  const float* Wkva = (const float*)d_in[3];
  const float* Wkvb = (const float*)d_in[4];
  const float* Wo   = (const float*)d_in[5];
  float* out = (float*)d_out;

  const int B = 2, N = 2048, M = B * N;   // M = 4096
  const int H = NUM_HEADS;

  char* ws = (char*)d_ws;
  auto alloc = [&](size_t halves) {
    _Float16* p = (_Float16*)ws;
    ws += (halves * sizeof(_Float16) + 255) & ~(size_t)255;
    return p;
  };
  _Float16* xh    = alloc((size_t)M * EMBED_DIM);
  _Float16* WqaT  = alloc((size_t)Q_LORA * EMBED_DIM);
  _Float16* WqbT  = alloc((size_t)(H * QH_DIM) * Q_LORA);
  _Float16* WkvaT = alloc((size_t)(KV_LORA + ROPE_DIM) * EMBED_DIM);
  _Float16* WkvbT = alloc((size_t)(H * 256) * KV_LORA);
  _Float16* WoT   = alloc((size_t)EMBED_DIM * EMBED_DIM);
  _Float16* qlora = alloc((size_t)M * Q_LORA);
  _Float16* qbuf  = alloc((size_t)M * H * QH_DIM);
  _Float16* ckv   = alloc((size_t)M * (KV_LORA + ROPE_DIM));
  _Float16* kv    = alloc((size_t)M * H * 256);
  _Float16* kf    = alloc((size_t)B * H * N * QH_DIM);
  _Float16* vt    = alloc((size_t)B * H * V_DIM * N);
  _Float16* ao    = alloc((size_t)M * EMBED_DIM);

  auto blocks = [](int n) { return (n + 255) / 256; };

  // --- precision conversion + weight transposition ---
  cvt_f32_f16<<<blocks(M * EMBED_DIM), 256, 0, stream>>>(x, xh, M * EMBED_DIM);
  cvt_transpose<<<blocks(EMBED_DIM * Q_LORA), 256, 0, stream>>>(
      Wqa, WqaT, EMBED_DIM, Q_LORA);
  cvt_transpose<<<blocks(Q_LORA * H * QH_DIM), 256, 0, stream>>>(
      Wqb, WqbT, Q_LORA, H * QH_DIM);
  cvt_transpose<<<blocks(EMBED_DIM * (KV_LORA + ROPE_DIM)), 256, 0, stream>>>(
      Wkva, WkvaT, EMBED_DIM, KV_LORA + ROPE_DIM);
  cvt_transpose<<<blocks(KV_LORA * H * 256), 256, 0, stream>>>(
      Wkvb, WkvbT, KV_LORA, H * 256);
  cvt_transpose<<<blocks(EMBED_DIM * EMBED_DIM), 256, 0, stream>>>(
      Wo, WoT, EMBED_DIM, EMBED_DIM);

  // --- projection GEMMs ---
  gemm_wmma<true><<<dim3(M / 128, Q_LORA / 64), 256, 0, stream>>>(
      xh, EMBED_DIM, WqaT, EMBED_DIM, qlora, Q_LORA, EMBED_DIM);
  gemm_wmma<true><<<dim3(M / 128, (H * QH_DIM) / 64), 256, 0, stream>>>(
      qlora, Q_LORA, WqbT, Q_LORA, qbuf, H * QH_DIM, Q_LORA);
  gemm_wmma<true><<<dim3(M / 128, (KV_LORA + ROPE_DIM) / 64), 256, 0, stream>>>(
      xh, EMBED_DIM, WkvaT, EMBED_DIM, ckv, KV_LORA + ROPE_DIM, EMBED_DIM);
  gemm_wmma<true><<<dim3(M / 128, (H * 256) / 64), 256, 0, stream>>>(
      ckv, KV_LORA + ROPE_DIM, WkvbT, KV_LORA, kv, H * 256, KV_LORA);

  // --- RoPE + head packing ---
  rope_q<<<blocks(M * H * 32), 256, 0, stream>>>(qbuf, M * H * 32, N);
  rope_k<<<blocks(M * 32), 256, 0, stream>>>(ckv, M * 32, N);
  pack_kv<<<dim3(N, H, B), 192, 0, stream>>>(kv, ckv, kf, vt, N);

  // --- causal attention ---
  mla_attn<<<dim3(N / 128, H, B), 256, 0, stream>>>(qbuf, kf, vt, ao, N);

  // --- output projection (fp32 out) ---
  gemm_wmma<false><<<dim3(M / 128, EMBED_DIM / 64), 256, 0, stream>>>(
      ao, EMBED_DIM, WoT, EMBED_DIM, out, EMBED_DIM, EMBED_DIM);
}
